// ReactionModel_21577915695446
// MI455X (gfx1250) — matrix operand order repaired and checked by reading, not compiled, not run on hardware
//
#include <hip/hip_runtime.h>
#include <hip/hip_bf16.h>

typedef __attribute__((ext_vector_type(16))) __bf16 v16bf;
typedef __attribute__((ext_vector_type(8)))  float  v8f;

#define HIDD 512
#define E_CNT 150000
#define NNODE 50000
#define NGRAPH 2000

__device__ __forceinline__ __bf16 f2b(float f) { return (__bf16)f; }

// ---------------- utility kernels ----------------

__global__ void fill_zero(float* __restrict__ p, long n) {
  long i = (long)blockIdx.x * blockDim.x + threadIdx.x;
  long stride = (long)gridDim.x * blockDim.x;
  for (; i < n; i += stride) p[i] = 0.f;
}

__global__ void f32_to_bf16(const float* __restrict__ in, __bf16* __restrict__ out, long n) {
  long i = (long)blockIdx.x * blockDim.x + threadIdx.x;
  long stride = (long)gridDim.x * blockDim.x;
  for (; i < n; i += stride) out[i] = f2b(in[i]);
}

// Repack f32 row-major weight [K,N] into bf16 WMMA B-fragment layout:
// fragment (32k x 16n) blocks, frag index = n16*(K/32)+k32, within a fragment
// lane L (0..31) holds 16 contiguous bf16: element e -> B[k32*32 + (L/16)*16 + e][n16*16 + (L%16)]
__global__ void repack_weight(const float* __restrict__ W, __bf16* __restrict__ out, int K, int N) {
  int i = blockIdx.x * blockDim.x + threadIdx.x;
  int total = K * N;
  if (i >= total) return;
  int frag = i >> 9;
  int r = i & 511;
  int lane = r >> 4, e = r & 15;
  int Kf = K >> 5;
  int n16 = frag / Kf, k32 = frag - n16 * Kf;
  int k = (k32 << 5) + ((lane >> 4) << 4) + e;
  int n = (n16 << 4) + (lane & 15);
  out[i] = f2b(W[(size_t)k * N + n]);
}

// small-K dense projection: out = act(X @ W + b) -> bf16
__global__ void proj_kernel(const float* __restrict__ X, const float* __restrict__ W,
                            const float* __restrict__ bias, __bf16* __restrict__ out,
                            long M, int Kin, int N, int relu) {
  long total = M * N;
  long stride = (long)gridDim.x * blockDim.x;
  for (long i = (long)blockIdx.x * blockDim.x + threadIdx.x; i < total; i += stride) {
    long r = i / N;
    int n = (int)(i - r * N);
    float s = bias[n];
    const float* xr = X + r * Kin;
    for (int k = 0; k < Kin; ++k) s += xr[k] * W[(size_t)k * N + n];
    if (relu) s = s > 0.f ? s : 0.f;
    out[i] = f2b(s);
  }
}

__global__ void pool_kernel(const __bf16* __restrict__ h, const int* __restrict__ batch,
                            float* __restrict__ pooled) {
  long total = (long)NNODE * HIDD;
  long stride = (long)gridDim.x * blockDim.x;
  for (long i = (long)blockIdx.x * blockDim.x + threadIdx.x; i < total; i += stride) {
    long r = i >> 9;
    int n = (int)(i & 511);
    atomicAdd(&pooled[(size_t)batch[r] * HIDD + n], (float)h[i]);
  }
}

__global__ void final_dot(const __bf16* __restrict__ p, const float* __restrict__ W,
                          const float* __restrict__ b, float* __restrict__ out, int G, int K) {
  int g = blockIdx.x * blockDim.x + threadIdx.x;
  if (g >= G) return;
  float s = b[0];
  const __bf16* pr = p + (size_t)g * K;
  for (int k = 0; k < K; ++k) s += (float)pr[k] * W[k];
  out[g] = s;
}

// ---------------- main WMMA GEMM ----------------
// AMODE 0: A = bf16 [M,K] row-major
// AMODE 1: A row r = concat( hb[src[r]] (512), eb[r] (512) )           (msg GEMM)
// AMODE 2: A row r = concat( hb[r] (512), cvt_bf16(agg_f32[r]) (512) ) (update1 GEMM)
// EPI 0: Cb[gm,N] = (relu?) (acc + bias)           -> bf16
// EPI 1: atomicAdd Cf[dst[gm]*N + gn] += relu(acc + bias)  (fused scatter)
// EPI 2: Cb = prelu(acc + bias, alpha)             -> bf16
template <int AMODE, int EPI>
__global__ __launch_bounds__(256) void wmma_gemm(
    const __bf16* __restrict__ A, const __bf16* __restrict__ A2,
    const float* __restrict__ A2f, const int* __restrict__ srcIdx,
    const int* __restrict__ dstIdx, const __bf16* __restrict__ Bfrag,
    const float* __restrict__ bias, const float* __restrict__ alphaP,
    float* __restrict__ Cf, __bf16* __restrict__ Cb,
    int M, int N, int K, int doRelu) {
  __shared__ __bf16 sA[16 * (1024 + 8)];  // +8 elem pad per row: avoids LDS bank conflicts
  const int LDA = K + 8;
  const int mb = blockIdx.x * 16;
  const int nb0 = blockIdx.y * 128;
  const int tid = threadIdx.x;

  // ---- cooperative A-tile staging (16 rows x K) into LDS, 16B chunks ----
  const int chunksPerRow = K >> 3;
  const int totalChunks = chunksPerRow << 4;
  for (int c = tid; c < totalChunks; c += 256) {
    int row = c / chunksPerRow;
    int ko = (c - row * chunksPerRow) << 3;
    uint4 val;
    if (AMODE == 0) {
      val = *(const uint4*)(A + (size_t)(mb + row) * K + ko);
    } else if (AMODE == 1) {
      int r = mb + row;
      if (ko < HIDD) {
        int g = srcIdx[r];
        val = *(const uint4*)(A + (size_t)g * HIDD + ko);
      } else {
        val = *(const uint4*)(A2 + (size_t)r * HIDD + (ko - HIDD));
      }
    } else {
      int r = mb + row;
      if (ko < HIDD) {
        val = *(const uint4*)(A + (size_t)r * HIDD + ko);
      } else {
        const float* pf = A2f + (size_t)r * HIDD + (ko - HIDD);
        union { __bf16 b[8]; uint4 u; } cv;
#pragma unroll
        for (int j = 0; j < 8; ++j) cv.b[j] = f2b(pf[j]);
        val = cv.u;
      }
    }
    *(uint4*)&sA[row * LDA + ko] = val;
  }
  __syncthreads();

  const int wave = tid >> 5;
  const int lane = tid & 31;
  const int nb = nb0 + wave * 16;  // this wave's 16-col output slice

  // A fragment source in LDS: row = lane%16, K-half selected by lane/16
  const int ar = lane & 15;
  const int khalf = lane >> 4;
  const __bf16* sArow = sA + ar * LDA + khalf * 16;

  // B fragments: contiguous 512 bf16 per fragment, lane gets 16 contiguous
  const size_t fragBase =
      ((size_t)(nb >> 4) * (size_t)(K >> 5)) * 512 + (size_t)lane * 16;

  v8f acc = {};
  const int nK = K >> 5;
  for (int ki = 0; ki < nK; ++ki) {
    union { uint4 u[2]; v16bf v; } af, bfv;
    const __bf16* ap = sArow + ki * 32;
    af.u[0] = *(const uint4*)(ap);        // K = base .. base+7
    af.u[1] = *(const uint4*)(ap + 8);    // K = base+8 .. base+15
    const __bf16* bp = Bfrag + fragBase + (size_t)ki * 512;
    bfv.u[0] = *(const uint4*)(bp);
    bfv.u[1] = *(const uint4*)(bp + 8);
    // next B fragment: locality 3 -> WGP-scope prefetch (pull into all cache levels),
    // not SYS scope (which skips the WGP cache entirely)
    __builtin_prefetch(bp + 512, 0, 3);
    acc = __builtin_amdgcn_wmma_f32_16x16x32_bf16(false, af.v, false, bfv.v,
                                                  (short)0, acc, false, false);
  }

  // ---- epilogue: C layout VGPR v, lanes 0-15 -> M=v, lanes 16-31 -> M=v+8, N=lane%16
  const int col = lane & 15;
  const int rsel = lane >> 4;
  const int gn = nb + col;
  const float bv = bias[gn];
  const float al = (EPI == 2) ? alphaP[0] : 0.f;
#pragma unroll
  for (int v = 0; v < 8; ++v) {
    int gm = mb + v + rsel * 8;
    float x = acc[v] + bv;
    if (EPI == 1) {
      x = x > 0.f ? x : 0.f;
      atomicAdd(Cf + (size_t)dstIdx[gm] * N + gn, x);
    } else if (EPI == 0) {
      if (doRelu) x = x > 0.f ? x : 0.f;
      Cb[(size_t)gm * N + gn] = f2b(x);
    } else {
      x = x >= 0.f ? x : al * x;
      Cb[(size_t)gm * N + gn] = f2b(x);
    }
  }
}

// ---------------- host launcher ----------------

extern "C" void kernel_launch(void* const* d_in, const int* in_sizes, int n_in,
                              void* d_out, int out_size, void* d_ws, size_t ws_size,
                              hipStream_t stream) {
  (void)in_sizes; (void)n_in; (void)out_size; (void)ws_size;
  const float* x         = (const float*)d_in[0];
  const float* edge_attr = (const float*)d_in[1];
  const int*   edge_index= (const int*)d_in[2];
  const int*   batch     = (const int*)d_in[3];
  const float* W_node = (const float*)d_in[4];
  const float* b_node = (const float*)d_in[5];
  const float* W_edge = (const float*)d_in[6];
  const float* b_edge = (const float*)d_in[7];
  const float* Wm  = (const float*)d_in[8];
  const float* bm  = (const float*)d_in[9];
  const float* Wu1 = (const float*)d_in[10];
  const float* bu1 = (const float*)d_in[11];
  const float* Wu2 = (const float*)d_in[12];
  const float* bu2 = (const float*)d_in[13];
  const float* Ws  = (const float*)d_in[14];
  const float* bs  = (const float*)d_in[15];
  const float* a_s = (const float*)d_in[16];
  const float* Wp1 = (const float*)d_in[17];
  const float* bp1 = (const float*)d_in[18];
  const float* a1  = (const float*)d_in[19];
  const float* Wp2 = (const float*)d_in[20];
  const float* bp2 = (const float*)d_in[21];
  const float* a2  = (const float*)d_in[22];
  const float* Wp3 = (const float*)d_in[23];
  const float* bp3 = (const float*)d_in[24];
  float* out = (float*)d_out;

  char* ws = (char*)d_ws;
  size_t off = 0;
  auto alloc = [&](size_t bytes) -> char* {
    char* p = ws + off;
    off += (bytes + 255) & ~(size_t)255;
    return p;
  };
  __bf16* hb      = (__bf16*)alloc((size_t)NNODE * HIDD * 2);
  __bf16* eb      = (__bf16*)alloc((size_t)E_CNT * HIDD * 2);
  float*  agg     = (float*) alloc((size_t)NNODE * HIDD * 4);
  __bf16* ub      = (__bf16*)alloc((size_t)NNODE * 1024 * 2);
  float*  pooled  = (float*) alloc((size_t)NGRAPH * HIDD * 4);
  __bf16* pooledb = (__bf16*)alloc((size_t)NGRAPH * HIDD * 2);
  __bf16* rb      = (__bf16*)alloc((size_t)NGRAPH * 1024 * 2);
  __bf16* p1b     = (__bf16*)alloc((size_t)NGRAPH * 512 * 2);
  __bf16* p2b     = (__bf16*)alloc((size_t)NGRAPH * 512 * 2);
  __bf16* WmB     = (__bf16*)alloc((size_t)5 * 1024 * 512 * 2);
  __bf16* Wu1B    = (__bf16*)alloc((size_t)5 * 1024 * 1024 * 2);
  __bf16* Wu2B    = (__bf16*)alloc((size_t)5 * 1024 * 512 * 2);
  __bf16* WsB     = (__bf16*)alloc((size_t)512 * 1024 * 2);
  __bf16* Wp1B    = (__bf16*)alloc((size_t)1024 * 512 * 2);
  __bf16* Wp2B    = (__bf16*)alloc((size_t)512 * 512 * 2);

  auto repack = [&](const float* W, __bf16* o, int K, int N) {
    int total = K * N;
    repack_weight<<<(total + 255) / 256, 256, 0, stream>>>(W, o, K, N);
  };
  for (int i = 0; i < 5; ++i) {
    repack(Wm  + (size_t)i * 1024 * 512,  WmB  + (size_t)i * 1024 * 512,  1024, 512);
    repack(Wu1 + (size_t)i * 1024 * 1024, Wu1B + (size_t)i * 1024 * 1024, 1024, 1024);
    repack(Wu2 + (size_t)i * 1024 * 512,  Wu2B + (size_t)i * 1024 * 512,  1024, 512);
  }
  repack(Ws, WsB, 512, 1024);
  repack(Wp1, Wp1B, 1024, 512);
  repack(Wp2, Wp2B, 512, 512);

  // input projections (tiny K)
  {
    long tot = (long)NNODE * HIDD;
    proj_kernel<<<(int)((tot + 255) / 256), 256, 0, stream>>>(x, W_node, b_node, hb,
                                                              NNODE, 74, HIDD, 1);
    tot = (long)E_CNT * HIDD;
    proj_kernel<<<(int)((tot + 255) / 256), 256, 0, stream>>>(edge_attr, W_edge, b_edge,
                                                              eb, E_CNT, 12, HIDD, 0);
  }

  const int* srcI = edge_index;
  const int* dstI = edge_index + E_CNT;
  for (int i = 0; i < 5; ++i) {
    fill_zero<<<4096, 256, 0, stream>>>(agg, (long)NNODE * HIDD);
    // msg = relu([h[src]|e] @ Wm + bm), fused scatter-add into agg
    wmma_gemm<1, 1><<<dim3(E_CNT / 16, 512 / 128), 256, 0, stream>>>(
        hb, eb, nullptr, srcI, dstI, WmB + (size_t)i * 1024 * 512,
        bm + (size_t)i * 512, nullptr, agg, nullptr, E_CNT, 512, 1024, 0);
    // u = relu([h|agg] @ Wu1 + bu1)
    wmma_gemm<2, 0><<<dim3(NNODE / 16, 1024 / 128), 256, 0, stream>>>(
        hb, nullptr, agg, nullptr, nullptr, Wu1B + (size_t)i * 1024 * 1024,
        bu1 + (size_t)i * 1024, nullptr, nullptr, ub, NNODE, 1024, 1024, 1);
    // h = (relu?)(u @ Wu2 + bu2)
    wmma_gemm<0, 0><<<dim3(NNODE / 16, 512 / 128), 256, 0, stream>>>(
        ub, nullptr, nullptr, nullptr, nullptr, Wu2B + (size_t)i * 1024 * 512,
        bu2 + (size_t)i * 512, nullptr, nullptr, hb, NNODE, 512, 1024,
        (i < 4) ? 1 : 0);
  }

  fill_zero<<<1024, 256, 0, stream>>>(pooled, (long)NGRAPH * HIDD);
  pool_kernel<<<4096, 256, 0, stream>>>(hb, batch, pooled);
  f32_to_bf16<<<1024, 256, 0, stream>>>(pooled, pooledb, (long)NGRAPH * HIDD);

  // readout head
  wmma_gemm<0, 2><<<dim3(NGRAPH / 16, 1024 / 128), 256, 0, stream>>>(
      pooledb, nullptr, nullptr, nullptr, nullptr, WsB, bs, a_s, nullptr, rb,
      NGRAPH, 1024, 512, 0);
  wmma_gemm<0, 2><<<dim3(NGRAPH / 16, 512 / 128), 256, 0, stream>>>(
      rb, nullptr, nullptr, nullptr, nullptr, Wp1B, bp1, a1, nullptr, p1b,
      NGRAPH, 512, 1024, 0);
  wmma_gemm<0, 2><<<dim3(NGRAPH / 16, 512 / 128), 256, 0, stream>>>(
      p1b, nullptr, nullptr, nullptr, nullptr, Wp2B, bp2, a2, nullptr, p2b,
      NGRAPH, 512, 512, 0);
  final_dot<<<(NGRAPH + 255) / 256, 256, 0, stream>>>(p2b, Wp3, bp3, out, NGRAPH, 512);
}
